// LSTMModel_86380382257305
// MI455X (gfx1250) — matrix operand order repaired and checked by reading, not compiled, not run on hardware
//
#include <hip/hip_runtime.h>
#include <hip/hip_bf16.h>

#define B_  32
#define S_  128
#define T_  128
#define E_  512
#define H_  512
#define G4  2048   // 4*H
#define V_  32000

typedef __attribute__((ext_vector_type(2)))  float     v2f;
typedef __attribute__((ext_vector_type(8)))  float     v8f;
typedef __attribute__((ext_vector_type(8)))  _Float16  v8h;
typedef __attribute__((ext_vector_type(16))) _Float16  v16h;

// ---------------------------------------------------------------------------
// Embedding gather: out[(s*B+b)*512 + e] = emb[idx[b*L+s]*512 + e]
// ---------------------------------------------------------------------------
__global__ void embed_kernel(float* __restrict__ out, const float* __restrict__ emb,
                             const int* __restrict__ idx, int L) {
    int total = L * B_ * 512;
    for (int i = blockIdx.x * blockDim.x + threadIdx.x; i < total;
         i += gridDim.x * blockDim.x) {
        int e = i & 511;
        int r = i >> 9;          // r = s*B + b
        int s = r >> 5;          // B_ == 32
        int b = r & 31;
        out[i] = emb[idx[b * L + s] * 512 + e];
    }
}

// ---------------------------------------------------------------------------
// bias sum: out[i] = a[i] + b[i]
// ---------------------------------------------------------------------------
__global__ void bias_sum_kernel(float* __restrict__ out, const float* __restrict__ a,
                                const float* __restrict__ b, int n) {
    int i = blockIdx.x * blockDim.x + threadIdx.x;
    if (i < n) out[i] = a[i] + b[i];
}

// ---------------------------------------------------------------------------
// f32 -> f16 convert
// ---------------------------------------------------------------------------
__global__ void to_half_kernel(_Float16* __restrict__ dst, const float* __restrict__ src,
                               int n) {
    for (int i = blockIdx.x * blockDim.x + threadIdx.x; i < n;
         i += gridDim.x * blockDim.x) {
        dst[i] = (_Float16)src[i];
    }
}

// ---------------------------------------------------------------------------
// Generic f32 WMMA GEMM: C[M,N] = A[M,K] @ W[N,K]^T (+ b1 + b2 broadcast on N)
// block = 256 thr (8 waves), wave -> one 16x16 tile; grid (N/128, M/16)
// ---------------------------------------------------------------------------
__global__ void __launch_bounds__(256)
gemm_wmma_f32(const float* __restrict__ A, int lda,
              const float* __restrict__ W, int ldw,
              float* __restrict__ C, int ldc, int K,
              const float* __restrict__ b1, const float* __restrict__ b2) {
    int tid = threadIdx.x, wave = tid >> 5, lane = tid & 31;
    int l = lane & 15, half = lane >> 4;
    int n0 = blockIdx.x * 128 + wave * 16;
    int m0 = blockIdx.y * 16;

    v8f acc = {};
    if (b1) {
        float bv = b1[n0 + l] + (b2 ? b2[n0 + l] : 0.0f);
        #pragma unroll
        for (int r = 0; r < 8; ++r) acc[r] = bv;
    }
    const float* Arow = A + (size_t)(m0 + l) * lda;
    const float* Wrow = W + (size_t)(n0 + l) * ldw;
    for (int k = 0; k < K; k += 4) {
        int kk = k + 2 * half;
        v2f a = { Arow[kk], Arow[kk + 1] };
        v2f w = { Wrow[kk], Wrow[kk + 1] };
        acc = __builtin_amdgcn_wmma_f32_16x16x4_f32(false, a, false, w,
                                                    (short)0, acc, false, false);
    }
    #pragma unroll
    for (int r = 0; r < 8; ++r)
        C[(size_t)(m0 + r + 8 * half) * ldc + n0 + l] = acc[r];
}

// ---------------------------------------------------------------------------
// Fused LSTM step: gates = Xp_t (or bsum) + [A2 @ W2^T] + hprev @ Whh^T
// then pointwise LSTM cell. 4 waves: one 16x16 gate tile each (i,f,g,o).
// grid (H/16, B/16), block 128.
// ---------------------------------------------------------------------------
__global__ void __launch_bounds__(128)
lstm_step_kernel(const float* __restrict__ Xp,    // [B, 4H] pre-gates (or null)
                 const float* __restrict__ bsum,  // [4H] bias sum (or null)
                 const float* __restrict__ A2,    // [B, K2] extra input (or null)
                 const float* __restrict__ W2, int ld2, int K2,
                 const float* __restrict__ hprev, // [B, H]
                 const float* __restrict__ Whh, int ldw,
                 float* __restrict__ hnext,       // [B, H]
                 float* __restrict__ c,           // [B, H] in-place
                 float* __restrict__ outseq)      // [B, H] slice (or null)
{
    int tid = threadIdx.x, wave = tid >> 5, lane = tid & 31;
    int l = lane & 15, half = lane >> 4;
    int j0 = blockIdx.x * 16;     // unit within H
    int b0 = blockIdx.y * 16;     // batch
    int col = wave * 512 + j0;    // gate column base in [0, 4H)

    v8f acc = {};
    if (Xp) {
        #pragma unroll
        for (int r = 0; r < 8; ++r)
            acc[r] = Xp[(size_t)(b0 + r + 8 * half) * G4 + col + l];
    } else if (bsum) {
        float bv = bsum[col + l];
        #pragma unroll
        for (int r = 0; r < 8; ++r) acc[r] = bv;
    }

    if (A2) {
        const float* Arow = A2 + (size_t)(b0 + l) * K2;
        const float* Wrow = W2 + (size_t)(col + l) * ld2;
        for (int k = 0; k < K2; k += 4) {
            int kk = k + 2 * half;
            v2f a = { Arow[kk], Arow[kk + 1] };
            v2f w = { Wrow[kk], Wrow[kk + 1] };
            acc = __builtin_amdgcn_wmma_f32_16x16x4_f32(false, a, false, w,
                                                        (short)0, acc, false, false);
        }
    }
    {
        const float* Arow = hprev + (size_t)(b0 + l) * H_;
        const float* Wrow = Whh + (size_t)(col + l) * ldw;
        for (int k = 0; k < H_; k += 4) {
            int kk = k + 2 * half;
            v2f a = { Arow[kk], Arow[kk + 1] };
            v2f w = { Wrow[kk], Wrow[kk + 1] };
            acc = __builtin_amdgcn_wmma_f32_16x16x4_f32(false, a, false, w,
                                                        (short)0, acc, false, false);
        }
    }

    __shared__ float g[4][16][16];
    #pragma unroll
    for (int r = 0; r < 8; ++r) g[wave][r + 8 * half][l] = acc[r];
    __syncthreads();

    for (int e = tid; e < 256; e += 128) {
        int m = e >> 4, j = e & 15;
        float iv = g[0][m][j], fv = g[1][m][j], gv = g[2][m][j], ov = g[3][m][j];
        int cidx = (b0 + m) * H_ + j0 + j;
        float cp = c[cidx];
        float ig = 1.0f / (1.0f + expf(-iv));
        float fg = 1.0f / (1.0f + expf(-fv));
        float og = 1.0f / (1.0f + expf(-ov));
        float cn = fg * cp + ig * tanhf(gv);
        float hn = og * tanhf(cn);
        c[cidx] = cn;
        hnext[cidx] = hn;
        if (outseq) outseq[cidx] = hn;
    }
}

// ---------------------------------------------------------------------------
// Attention: per batch element b — scores over S, softmax, weighted context.
// grid 32, block 128.
// ---------------------------------------------------------------------------
__global__ void __launch_bounds__(128)
attention_kernel(const float* __restrict__ enc,   // [S, B, H]
                 const float* __restrict__ q,     // [B, H]
                 float* __restrict__ ctx)         // [B, H]
{
    __shared__ float sq[H_];
    __shared__ float sa[S_];
    __shared__ float red[128];
    int b = blockIdx.x, tid = threadIdx.x;

    for (int k = tid; k < H_; k += 128) sq[k] = q[b * H_ + k];
    __syncthreads();

    // one score per thread (S_ == 128 == blockDim)
    const float* row = enc + (size_t)(tid * B_ + b) * H_;
    float sc = 0.0f;
    for (int k = 0; k < H_; ++k) sc += row[k] * sq[k];

    red[tid] = sc; __syncthreads();
    for (int st = 64; st > 0; st >>= 1) {
        if (tid < st) red[tid] = fmaxf(red[tid], red[tid + st]);
        __syncthreads();
    }
    float mx = red[0]; __syncthreads();
    float ex = expf(sc - mx);
    red[tid] = ex; __syncthreads();
    for (int st = 64; st > 0; st >>= 1) {
        if (tid < st) red[tid] += red[tid + st];
        __syncthreads();
    }
    float inv = 1.0f / red[0]; __syncthreads();
    sa[tid] = ex * inv;
    __syncthreads();

    for (int k = tid; k < H_; k += 128) {
        float cv = 0.0f;
        for (int s = 0; s < S_; ++s) cv += sa[s] * enc[(size_t)(s * B_ + b) * H_ + k];
        ctx[b * H_ + k] = cv;
    }
}

// ---------------------------------------------------------------------------
// attn out: out = tanh([ctx, h1] @ Wout^T); store y[B,T,H] slice + feed.
// grid (4, 2), block 256 (8 waves -> 8 N-tiles).
// ---------------------------------------------------------------------------
__global__ void __launch_bounds__(256)
attn_out_kernel(const float* __restrict__ ctx, const float* __restrict__ h1,
                const float* __restrict__ Wout,   // [H, 2H]
                float* __restrict__ y, float* __restrict__ feed, int t) {
    int tid = threadIdx.x, wave = tid >> 5, lane = tid & 31;
    int l = lane & 15, half = lane >> 4;
    int n0 = blockIdx.x * 128 + wave * 16;
    int m0 = blockIdx.y * 16;

    v8f acc = {};
    const float* WrowA = Wout + (size_t)(n0 + l) * (2 * H_);
    const float* WrowB = WrowA + H_;
    const float* Crow  = ctx + (size_t)(m0 + l) * H_;
    const float* Hrow  = h1  + (size_t)(m0 + l) * H_;
    for (int k = 0; k < H_; k += 4) {
        int kk = k + 2 * half;
        v2f a = { Crow[kk], Crow[kk + 1] };
        v2f w = { WrowA[kk], WrowA[kk + 1] };
        acc = __builtin_amdgcn_wmma_f32_16x16x4_f32(false, a, false, w,
                                                    (short)0, acc, false, false);
    }
    for (int k = 0; k < H_; k += 4) {
        int kk = k + 2 * half;
        v2f a = { Hrow[kk], Hrow[kk + 1] };
        v2f w = { WrowB[kk], WrowB[kk + 1] };
        acc = __builtin_amdgcn_wmma_f32_16x16x4_f32(false, a, false, w,
                                                    (short)0, acc, false, false);
    }
    #pragma unroll
    for (int r = 0; r < 8; ++r) {
        float v = tanhf(acc[r]);
        int b = m0 + r + 8 * half;
        int n = n0 + l;
        y[(size_t)(b * T_ + t) * H_ + n] = v;   // [B, T, H]
        feed[b * H_ + n] = v;
    }
}

// ---------------------------------------------------------------------------
// Final projection (f16 WMMA, f32 accumulate): out[M,V] = A[M,512] @ W[V,512]^T + bias
// grid (V/128, M/16), block 256 (8 waves -> 8 N-tiles).
// ---------------------------------------------------------------------------
__global__ void __launch_bounds__(256)
fc_wmma_f16(const _Float16* __restrict__ Ah, const _Float16* __restrict__ Wh,
            const float* __restrict__ bias, float* __restrict__ out) {
    int tid = threadIdx.x, wave = tid >> 5, lane = tid & 31;
    int l = lane & 15, half = lane >> 4;
    int n0 = blockIdx.x * 128 + wave * 16;
    int m0 = blockIdx.y * 16;

    v8f acc = {};
    const _Float16* arow = Ah + (size_t)(m0 + l) * 512 + half * 16;
    const _Float16* wrow = Wh + (size_t)(n0 + l) * 512 + half * 16;
    for (int k = 0; k < 512; k += 32) {
        v8h alo = *(const v8h*)(arow + k);
        v8h ahi = *(const v8h*)(arow + k + 8);
        v16h a = __builtin_shufflevector(alo, ahi, 0, 1, 2, 3, 4, 5, 6, 7,
                                         8, 9, 10, 11, 12, 13, 14, 15);
        v16h b = *(const v16h*)(wrow + k);
        acc = __builtin_amdgcn_wmma_f32_16x16x32_f16(false, a, false, b,
                                                     (short)0, acc, false, false);
    }
    float bv = bias[n0 + l];
    #pragma unroll
    for (int r = 0; r < 8; ++r)
        out[(size_t)(m0 + r + 8 * half) * V_ + n0 + l] = acc[r] + bv;
}

// ---------------------------------------------------------------------------
// Host orchestration
// ---------------------------------------------------------------------------
extern "C" void kernel_launch(void* const* d_in, const int* in_sizes, int n_in,
                              void* d_out, int out_size, void* d_ws, size_t ws_size,
                              hipStream_t stream) {
    (void)in_sizes; (void)n_in; (void)out_size; (void)ws_size;

    const int*   src     = (const int*)d_in[0];
    const int*   trg     = (const int*)d_in[1];
    const float* enc_emb = (const float*)d_in[2];
    const float* dec_emb = (const float*)d_in[3];
    const float* e_Wih0  = (const float*)d_in[4];
    const float* e_Whh0  = (const float*)d_in[5];
    const float* e_bih0  = (const float*)d_in[6];
    const float* e_bhh0  = (const float*)d_in[7];
    const float* e_Wih1  = (const float*)d_in[8];
    const float* e_Whh1  = (const float*)d_in[9];
    const float* e_bih1  = (const float*)d_in[10];
    const float* e_bhh1  = (const float*)d_in[11];
    const float* d_Wih0  = (const float*)d_in[12];
    const float* d_Whh0  = (const float*)d_in[13];
    const float* d_bih0  = (const float*)d_in[14];
    const float* d_bhh0  = (const float*)d_in[15];
    const float* d_Wih1  = (const float*)d_in[16];
    const float* d_Whh1  = (const float*)d_in[17];
    const float* d_bih1  = (const float*)d_in[18];
    const float* d_bhh1  = (const float*)d_in[19];
    const float* att_Win = (const float*)d_in[20];
    const float* att_Wout= (const float*)d_in[21];
    const float* fc_W    = (const float*)d_in[22];
    const float* fc_b    = (const float*)d_in[23];
    float* out = (float*)d_out;

    // ---- workspace carve-out ----
    char* w = (char*)d_ws;
    auto carve = [&](size_t bytes) -> void* {
        void* p = (void*)w;
        w += (bytes + 255) & ~(size_t)255;
        return p;
    };
    const size_t MB = (size_t)(S_ * B_);          // 4096 rows
    float* enc_x   = (float*)carve(MB * E_ * 4);
    float* Xp      = (float*)carve(MB * G4 * 4);  // reused: Xp0, Xp1, Xd0
    float* out0    = (float*)carve(MB * H_ * 4);
    float* encouts = (float*)carve(MB * H_ * 4);
    float* dec_x   = (float*)carve(MB * E_ * 4);
    float* yb      = (float*)carve(MB * H_ * 4);  // [B, T, H]
    float* h0a     = (float*)carve(B_ * H_ * 4);
    float* h0b     = (float*)carve(B_ * H_ * 4);
    float* h1a     = (float*)carve(B_ * H_ * 4);
    float* h1b     = (float*)carve(B_ * H_ * 4);
    float* c0      = (float*)carve(B_ * H_ * 4);
    float* c1      = (float*)carve(B_ * H_ * 4);
    float* qb      = (float*)carve(B_ * H_ * 4);
    float* ctx     = (float*)carve(B_ * H_ * 4);
    float* feed    = (float*)carve(B_ * H_ * 4);
    float* bsum1   = (float*)carve(G4 * 4);
    _Float16* y_h  = (_Float16*)carve(MB * H_ * 2);
    _Float16* W_h  = (_Float16*)carve((size_t)V_ * H_ * 2);

    // zero initial states
    hipMemsetAsync(h0a, 0, B_ * H_ * 4, stream);
    hipMemsetAsync(h1a, 0, B_ * H_ * 4, stream);
    hipMemsetAsync(c0,  0, B_ * H_ * 4, stream);
    hipMemsetAsync(c1,  0, B_ * H_ * 4, stream);
    hipMemsetAsync(feed,0, B_ * H_ * 4, stream);

    dim3 stepGrid(H_ / 16, B_ / 16);     // (32, 2)
    dim3 bigGrid(G4 / 128, MB / 16);     // (16, 256)
    dim3 smallGrid(H_ / 128, B_ / 16);   // (4, 2)

    // ================= encoder =================
    embed_kernel<<<1024, 256, 0, stream>>>(enc_x, enc_emb, src, S_);

    // layer 0: pre-gates then recurrence
    gemm_wmma_f32<<<bigGrid, 256, 0, stream>>>(enc_x, E_, e_Wih0, E_, Xp, G4, E_,
                                               e_bih0, e_bhh0);
    float* hc = h0a; float* hn = h0b;
    for (int t = 0; t < S_; ++t) {
        lstm_step_kernel<<<stepGrid, 128, 0, stream>>>(
            Xp + (size_t)t * B_ * G4, nullptr, nullptr, nullptr, 0, 0,
            hc, e_Whh0, H_, hn, c0, out0 + (size_t)t * B_ * H_);
        float* tmp = hc; hc = hn; hn = tmp;
    }
    // layer 1
    gemm_wmma_f32<<<bigGrid, 256, 0, stream>>>(out0, H_, e_Wih1, H_, Xp, G4, H_,
                                               e_bih1, e_bhh1);
    float* hc1 = h1a; float* hn1 = h1b;
    for (int t = 0; t < S_; ++t) {
        lstm_step_kernel<<<stepGrid, 128, 0, stream>>>(
            Xp + (size_t)t * B_ * G4, nullptr, nullptr, nullptr, 0, 0,
            hc1, e_Whh1, H_, hn1, c1, encouts + (size_t)t * B_ * H_);
        float* tmp = hc1; hc1 = hn1; hn1 = tmp;
    }
    // hc / hc1 now hold encoder final h; c0 / c1 hold final c (decoder init)

    // ================= decoder =================
    embed_kernel<<<1024, 256, 0, stream>>>(dec_x, dec_emb, trg, T_);
    // pre-gates from embedded input (first E columns of d_Wih0, row stride E+H)
    gemm_wmma_f32<<<bigGrid, 256, 0, stream>>>(dec_x, E_, d_Wih0, E_ + H_, Xp, G4, E_,
                                               d_bih0, d_bhh0);
    bias_sum_kernel<<<G4 / 256, 256, 0, stream>>>(bsum1, d_bih1, d_bhh1, G4);

    for (int t = 0; t < T_; ++t) {
        // cell 0: Xd0[t] + feed @ Wih0[:,E:]^T + h0 @ Whh0^T
        lstm_step_kernel<<<stepGrid, 128, 0, stream>>>(
            Xp + (size_t)t * B_ * G4, nullptr,
            feed, d_Wih0 + E_, E_ + H_, H_,
            hc, d_Whh0, H_, hn, c0, nullptr);
        { float* tmp = hc; hc = hn; hn = tmp; }
        // cell 1: bias + h0 @ Wih1^T + h1 @ Whh1^T
        lstm_step_kernel<<<stepGrid, 128, 0, stream>>>(
            nullptr, bsum1,
            hc, d_Wih1, H_, H_,
            hc1, d_Whh1, H_, hn1, c1, nullptr);
        { float* tmp = hc1; hc1 = hn1; hn1 = tmp; }
        // q = h1 @ att_Win^T
        gemm_wmma_f32<<<smallGrid, 256, 0, stream>>>(hc1, H_, att_Win, H_,
                                                     qb, H_, H_, nullptr, nullptr);
        // attention softmax + context
        attention_kernel<<<B_, 128, 0, stream>>>(encouts, qb, ctx);
        // out = tanh([ctx, h1] @ att_Wout^T) -> y[:, t, :] and feed
        attn_out_kernel<<<smallGrid, 256, 0, stream>>>(ctx, hc1, att_Wout, yb, feed, t);
    }

    // ================= final projection =================
    to_half_kernel<<<2048, 256, 0, stream>>>(y_h, yb, (int)(MB * H_));
    to_half_kernel<<<4096, 256, 0, stream>>>(W_h, fc_W, V_ * H_);
    dim3 fcGrid(V_ / 128, MB / 16);      // (250, 256)
    fc_wmma_f16<<<fcGrid, 256, 0, stream>>>(y_h, W_h, fc_b, out);
}
